// RoPEAttention_82514911691426
// MI455X (gfx1250) — compile-verified, matrix operand order checked by reading, not run on hardware
//
#include <hip/hip_runtime.h>

typedef __attribute__((ext_vector_type(16))) __bf16 v16bf;
typedef __attribute__((ext_vector_type(8)))  __bf16 v8bf;
typedef __attribute__((ext_vector_type(8)))  float  v8f;
typedef __attribute__((ext_vector_type(4)))  unsigned int v4u;
typedef __attribute__((ext_vector_type(8)))  int   v8i;
typedef __attribute__((ext_vector_type(4)))  int   v4i;

#define D_MODEL 1024
#define N_HEADS 16
#define HEAD_DIM 64
#define SEQ 2048
#define BATCH 2
#define ROWS (BATCH * SEQ)   // 4096

// ---------------------------------------------------------------------------
// Tensor Data Mover: 2-D tile load, bf16 elements.
// D# packing per CDNA5 ISA:
//   group0: [1:0]=count=1, [63:32]=lds_addr, [120:64]=global_addr, [127:126]=2
//   group1: data_size=1 (2B), tensor_dim0/1, tile_dim0/1, tensor_dim0_stride
//   groups 2/3 (and trailing group): zero (<=2D tensor)
// ---------------------------------------------------------------------------
__device__ __forceinline__ unsigned int lds_off(const void* p) {
  // generic LDS address: low 32 bits are the LDS byte offset (aperture rule)
  return (unsigned int)(unsigned long long)(uintptr_t)p;
}

__device__ __forceinline__ void tdm_load_2d(unsigned int ldsaddr, const void* gaddr,
                                            unsigned int tensor_d0, unsigned int tensor_d1,
                                            unsigned int tile_d0, unsigned int tile_d1,
                                            unsigned int stride_d0) {
  unsigned long long ga = (unsigned long long)(uintptr_t)gaddr;
  v4u g0;
  g0[0] = 1u;                                       // count=1
  g0[1] = ldsaddr;                                  // lds_addr
  g0[2] = (unsigned int)(ga & 0xffffffffu);         // global_addr[31:0]
  g0[3] = (unsigned int)((ga >> 32) & 0x1ffffffu) | (2u << 30);  // [56:32] | type=2
  v8i g1;
  g1[0] = (int)(1u << 16);                                          // data_size=2B
  g1[1] = (int)((tensor_d0 & 0xffffu) << 16);                       // tensor_dim0 lo
  g1[2] = (int)(((tensor_d0 >> 16) & 0xffffu) | ((tensor_d1 & 0xffffu) << 16));
  g1[3] = (int)(((tensor_d1 >> 16) & 0xffffu) | ((tile_d0 & 0xffffu) << 16));
  g1[4] = (int)(tile_d1 & 0xffffu);                                 // tile_dim1, tile_dim2=0
  g1[5] = (int)stride_d0;                                           // dim0 stride lo
  g1[6] = 0;
  g1[7] = 0;
  v4i z4 = {0, 0, 0, 0};
  v8i z8 = {0, 0, 0, 0, 0, 0, 0, 0};
  __builtin_amdgcn_tensor_load_to_lds(g0, g1, z4, z4, z8, 0);
}

// ---------------------------------------------------------------------------
// Fragment loader: 16x32 bf16 tile (A-layout; B has identical per-lane pattern
// since both operands are K-contiguous in our layouts).
// Lane L: idx = L%16 (row for A / col for B), base = (L/16)*8.
// ---------------------------------------------------------------------------
__device__ __forceinline__ v16bf load_frag(const __bf16* p, int ld) {
  const int lane = threadIdx.x & 31;
  const int idx  = lane & 15;
  const int base = (lane >> 4) << 3;
  const __bf16* q = p + idx * ld + base;
  v8bf lo = *(const v8bf*)(q);
  v8bf hi = *(const v8bf*)(q + 16);
  v16bf f;
#pragma unroll
  for (int i = 0; i < 8; ++i) { f[i] = lo[i]; f[i + 8] = hi[i]; }
  return f;
}

__device__ __forceinline__ v8f wmma_bf16(v16bf a, v16bf b, v8f c) {
  return __builtin_amdgcn_wmma_f32_16x16x32_bf16(
      false, a, false, b, (short)0, c, false, false);
}

// ---------------------------------------------------------------------------
// fp32 -> bf16 convert
// ---------------------------------------------------------------------------
__global__ void f32_to_bf16(const float* __restrict__ in,
                            __bf16* __restrict__ out, int n) {
  int i = blockIdx.x * blockDim.x + threadIdx.x;
  if (i < n) out[i] = (__bf16)in[i];
}

// ---------------------------------------------------------------------------
// GEMM: C[4096,1024] f32 = A[4096,1024] bf16 @ Bw[1024,1024]^T bf16
// Block tile 128x64, 8 waves (wave tile 16x64). Double-buffered LDS staged by
// the Tensor Data Mover; TENSORcnt + barrier gates each stage.
// ---------------------------------------------------------------------------
__global__ __launch_bounds__(256) void gemm_bf16_nt(
    const __bf16* __restrict__ A, const __bf16* __restrict__ Bw,
    float* __restrict__ C) {
  __shared__ __bf16 As[2][128][32];   // 2 x 8KB
  __shared__ __bf16 Bs[2][64][32];    // 2 x 4KB
  const int wave = threadIdx.x >> 5;
  const int lane = threadIdx.x & 31;
  const int row0 = blockIdx.x * 128;
  const int col0 = blockIdx.y * 64;

  if (wave == 0) {
    tdm_load_2d(lds_off(&As[0][0][0]), A + (long)row0 * D_MODEL,
                D_MODEL, ROWS, 32, 128, D_MODEL);
    tdm_load_2d(lds_off(&Bs[0][0][0]), Bw + (long)col0 * D_MODEL,
                D_MODEL, D_MODEL, 32, 64, D_MODEL);
  }

  v8f acc[4] = {};
  const int NK = D_MODEL / 32;
  for (int i = 0; i < NK; ++i) {
    const int cur = i & 1;
    if (wave == 0) __builtin_amdgcn_s_wait_tensorcnt(0);
    __syncthreads();                       // stage `cur` visible to all waves
    if (wave == 0 && i + 1 < NK) {         // prefetch next stage via TDM
      const int nxt = cur ^ 1;
      const int k = (i + 1) * 32;
      tdm_load_2d(lds_off(&As[nxt][0][0]), A + (long)row0 * D_MODEL + k,
                  D_MODEL, ROWS, 32, 128, D_MODEL);
      tdm_load_2d(lds_off(&Bs[nxt][0][0]), Bw + (long)col0 * D_MODEL + k,
                  D_MODEL, D_MODEL, 32, 64, D_MODEL);
    }
    v16bf a = load_frag(&As[cur][wave * 16][0], 32);
#pragma unroll
    for (int t = 0; t < 4; ++t) {
      v16bf b = load_frag(&Bs[cur][t * 16][0], 32);
      acc[t] = wmma_bf16(a, b, acc[t]);
    }
    __syncthreads();                       // done reading before overwrite
  }

  const int n  = lane & 15;
  const int hi = lane >> 4;
  const int r0 = row0 + wave * 16;
#pragma unroll
  for (int t = 0; t < 4; ++t)
#pragma unroll
    for (int j = 0; j < 8; ++j)
      C[(long)(r0 + j + 8 * hi) * D_MODEL + col0 + t * 16 + n] = acc[t][j];
}

// ---------------------------------------------------------------------------
// RoPE on q,k (fp32 -> bf16, layout (B,S,H*64) kept) and V transposed to
// (B,H,64,S) bf16 so PV B-fragments are key-contiguous.
// ---------------------------------------------------------------------------
__global__ void rope_cvt(const float* __restrict__ qf,
                         const float* __restrict__ kf,
                         const float* __restrict__ vf,
                         __bf16* __restrict__ qb, __bf16* __restrict__ kb,
                         __bf16* __restrict__ vT) {
  int tid = blockIdx.x * blockDim.x + threadIdx.x;
  const int npairs = ROWS * D_MODEL / 2;
  if (tid >= npairs) return;
  int e    = tid * 2;
  int din  = e & 1023;
  int s    = (e >> 10) & 2047;
  int b    = e >> 21;
  int h    = din >> 6;
  int dpos = din & 63;
  float ang = (float)s * exp2f(-(float)dpos * (13.2877123795494f / 64.0f));
  float c = __cosf(ang), sn = __sinf(ang);

  float q0 = qf[e], q1 = qf[e + 1];
  qb[e]     = (__bf16)(q0 * c - q1 * sn);
  qb[e + 1] = (__bf16)(q0 * sn + q1 * c);
  float k0 = kf[e], k1 = kf[e + 1];
  kb[e]     = (__bf16)(k0 * c - k1 * sn);
  kb[e + 1] = (__bf16)(k0 * sn + k1 * c);

  long vbase = (((long)(b * N_HEADS + h) * HEAD_DIM) + dpos) * SEQ + s;
  vT[vbase]       = (__bf16)vf[e];
  vT[vbase + SEQ] = (__bf16)vf[e + 1];
}

// ---------------------------------------------------------------------------
// Flash attention: one block = one (b,h), 8 waves = 8 consecutive q-tiles of
// 16 rows. K/V tiles (32 keys) are TDM-staged into double-buffered LDS and
// shared by all 8 waves. Online softmax per 16-lane row group; P re-swizzled
// C-layout -> A-layout through per-wave LDS.
// ---------------------------------------------------------------------------
__global__ __launch_bounds__(256) void flash_attn(
    const __bf16* __restrict__ qb, const __bf16* __restrict__ kb,
    const __bf16* __restrict__ vT, __bf16* __restrict__ ob) {
  __shared__ __bf16 Ks[2][32][64];   // 32 keys x 64 dims     (2 x 4KB)
  __shared__ __bf16 Vs[2][64][32];   // 64 dims x 32 keys     (2 x 4KB)
  __shared__ __bf16 P[8][16][32];    // per-wave prob tile    (8KB)

  const int wl   = threadIdx.x >> 5;
  const int lane = threadIdx.x & 31;
  const int bh   = blockIdx.x >> 4;          // 512 blocks = 32 bh x 16 qblk
  const int qblk = blockIdx.x & 15;
  const int b    = bh >> 4, h = bh & 15;
  const int qt   = qblk * 8 + wl;            // q tile (16 rows each)
  const int n    = lane & 15, hi = lane >> 4;

  const __bf16* qptr  = qb + ((long)(b * SEQ + qt * 16)) * D_MODEL + h * HEAD_DIM;
  const __bf16* kbase = kb + ((long)(b * SEQ)) * D_MODEL + h * HEAD_DIM;
  const __bf16* vbase = vT + ((long)(b * N_HEADS + h) * HEAD_DIM) * SEQ;

  v16bf a0 = load_frag(qptr, D_MODEL);       // q d=0..31
  v16bf a1 = load_frag(qptr + 32, D_MODEL);  // q d=32..63

  v8f o0 = {}, o1 = {}, o2 = {}, o3 = {};
  float rmax[8], rsum[8];
#pragma unroll
  for (int j = 0; j < 8; ++j) { rmax[j] = -1e30f; rsum[j] = 0.f; }

  if (wl == 0) {
    tdm_load_2d(lds_off(&Ks[0][0][0]), kbase, D_MODEL, SEQ, 64, 32, D_MODEL);
    tdm_load_2d(lds_off(&Vs[0][0][0]), vbase, SEQ, HEAD_DIM, 32, 64, SEQ);
  }

  const int NKK = SEQ / 32;                  // 64 key chunks
  for (int i = 0; i < NKK; ++i) {
    const int cur = i & 1;
    if (wl == 0) __builtin_amdgcn_s_wait_tensorcnt(0);
    __syncthreads();
    if (wl == 0 && i + 1 < NKK) {
      const int nxt = cur ^ 1;
      const int kk = (i + 1) * 32;
      tdm_load_2d(lds_off(&Ks[nxt][0][0]), kbase + (long)kk * D_MODEL,
                  D_MODEL, SEQ, 64, 32, D_MODEL);
      tdm_load_2d(lds_off(&Vs[nxt][0][0]), vbase + kk,
                  SEQ, HEAD_DIM, 32, 64, SEQ);
    }

    // S = q @ k^T for 32 keys -> two 16x16 f32 fragments (from LDS K tile)
    v8f s0 = {}, s1 = {};
    s0 = wmma_bf16(a0, load_frag(&Ks[cur][0][0], 64), s0);
    s0 = wmma_bf16(a1, load_frag(&Ks[cur][0][32], 64), s0);
    s1 = wmma_bf16(a0, load_frag(&Ks[cur][16][0], 64), s1);
    s1 = wmma_bf16(a1, load_frag(&Ks[cur][16][32], 64), s1);

#pragma unroll
    for (int j = 0; j < 8; ++j) {
      float x0 = s0[j] * 0.125f;             // 1/sqrt(64)
      float x1 = s1[j] * 0.125f;
      float mc = fmaxf(x0, x1);
#pragma unroll
      for (int off = 8; off >= 1; off >>= 1)
        mc = fmaxf(mc, __shfl_xor(mc, off, 16));
      float mnew = fmaxf(rmax[j], mc);
      float corr = __expf(rmax[j] - mnew);
      rmax[j] = mnew;
      float p0 = __expf(x0 - mnew);
      float p1 = __expf(x1 - mnew);
      float ps = p0 + p1;
#pragma unroll
      for (int off = 8; off >= 1; off >>= 1)
        ps += __shfl_xor(ps, off, 16);
      rsum[j] = rsum[j] * corr + ps;
      o0[j] *= corr; o1[j] *= corr; o2[j] *= corr; o3[j] *= corr;
      P[wl][j + 8 * hi][n]      = (__bf16)p0;
      P[wl][j + 8 * hi][16 + n] = (__bf16)p1;
    }
    asm volatile("s_wait_dscnt 0" ::: "memory");   // per-wave DS in-order
    v16bf pa = load_frag(&P[wl][0][0], 32);

    // O += P @ V (V tile in LDS, d-major: rows = out dim, keys contiguous)
    o0 = wmma_bf16(pa, load_frag(&Vs[cur][0][0], 32), o0);
    o1 = wmma_bf16(pa, load_frag(&Vs[cur][16][0], 32), o1);
    o2 = wmma_bf16(pa, load_frag(&Vs[cur][32][0], 32), o2);
    o3 = wmma_bf16(pa, load_frag(&Vs[cur][48][0], 32), o3);

    __syncthreads();                          // done with stage before overwrite
  }

  __bf16* optr = ob + ((long)(b * SEQ + qt * 16)) * D_MODEL + h * HEAD_DIM;
#pragma unroll
  for (int j = 0; j < 8; ++j) {
    float inv = 1.0f / rsum[j];
    long r = j + 8 * hi;
    optr[r * D_MODEL + n]      = (__bf16)(o0[j] * inv);
    optr[r * D_MODEL + 16 + n] = (__bf16)(o1[j] * inv);
    optr[r * D_MODEL + 32 + n] = (__bf16)(o2[j] * inv);
    optr[r * D_MODEL + 48 + n] = (__bf16)(o3[j] * inv);
  }
}

// ---------------------------------------------------------------------------
// Launch
// ---------------------------------------------------------------------------
extern "C" void kernel_launch(void* const* d_in, const int* in_sizes, int n_in,
                              void* d_out, int out_size, void* d_ws, size_t ws_size,
                              hipStream_t stream) {
  (void)in_sizes; (void)n_in; (void)out_size; (void)ws_size;
  const float* x  = (const float*)d_in[0];
  const float* wq = (const float*)d_in[1];
  const float* wk = (const float*)d_in[2];
  const float* wv = (const float*)d_in[3];
  const float* wo = (const float*)d_in[4];

  char* ws = (char*)d_ws;
  size_t off = 0;
  __bf16* xb  = (__bf16*)(ws + off); off += (size_t)ROWS * D_MODEL * 2;
  __bf16* wqb = (__bf16*)(ws + off); off += (size_t)D_MODEL * D_MODEL * 2;
  __bf16* wkb = (__bf16*)(ws + off); off += (size_t)D_MODEL * D_MODEL * 2;
  __bf16* wvb = (__bf16*)(ws + off); off += (size_t)D_MODEL * D_MODEL * 2;
  __bf16* wob = (__bf16*)(ws + off); off += (size_t)D_MODEL * D_MODEL * 2;
  float*  qf  = (float*)(ws + off);  off += (size_t)ROWS * D_MODEL * 4;
  float*  kf  = (float*)(ws + off);  off += (size_t)ROWS * D_MODEL * 4;
  float*  vf  = (float*)(ws + off);  off += (size_t)ROWS * D_MODEL * 4;
  __bf16* qrb = (__bf16*)(ws + off); off += (size_t)ROWS * D_MODEL * 2;
  __bf16* krb = (__bf16*)(ws + off); off += (size_t)ROWS * D_MODEL * 2;
  __bf16* vTb = (__bf16*)(ws + off); off += (size_t)ROWS * D_MODEL * 2;
  __bf16* atb = (__bf16*)(ws + off); off += (size_t)ROWS * D_MODEL * 2;

  const int nx = ROWS * D_MODEL;
  const int nw = D_MODEL * D_MODEL;
  f32_to_bf16<<<nx / 256, 256, 0, stream>>>(x, xb, nx);
  f32_to_bf16<<<nw / 256, 256, 0, stream>>>(wq, wqb, nw);
  f32_to_bf16<<<nw / 256, 256, 0, stream>>>(wk, wkb, nw);
  f32_to_bf16<<<nw / 256, 256, 0, stream>>>(wv, wvb, nw);
  f32_to_bf16<<<nw / 256, 256, 0, stream>>>(wo, wob, nw);

  dim3 gg(ROWS / 128, D_MODEL / 64);    // (32,16)
  gemm_bf16_nt<<<gg, 256, 0, stream>>>(xb, wqb, qf);
  gemm_bf16_nt<<<gg, 256, 0, stream>>>(xb, wkb, kf);
  gemm_bf16_nt<<<gg, 256, 0, stream>>>(xb, wvb, vf);

  rope_cvt<<<(nx / 2) / 256, 256, 0, stream>>>(qf, kf, vf, qrb, krb, vTb);

  // 512 blocks: 32 (b,h) x 16 q-blocks of 128 rows
  flash_attn<<<BATCH * N_HEADS * (SEQ / 128), 256, 0, stream>>>(qrb, krb, vTb, atb);

  gemm_bf16_nt<<<gg, 256, 0, stream>>>(atb, wob, (float*)d_out);
}
